// PCGTConvLayer_31817117728798
// MI455X (gfx1250) — matrix operand order repaired and checked by reading, not compiled, not run on hardware
//
#include <hip/hip_runtime.h>
#include <hip/hip_bf16.h>
#include <cstdint>

// ---------------------------------------------------------------------------
// Types / helpers
// ---------------------------------------------------------------------------
typedef __attribute__((ext_vector_type(16))) __bf16 v16bf;
typedef __attribute__((ext_vector_type(8)))  float  v8f;

#define NROWS 65536
#define INC   256
#define NHEAD 4
#define DHEAD 64

__device__ __forceinline__ unsigned short f2bfu(float f) {
    unsigned u = __builtin_bit_cast(unsigned, f);
    u += 0x7FFFu + ((u >> 16) & 1u);            // round-to-nearest-even
    return (unsigned short)(u >> 16);
}
__device__ __forceinline__ __bf16 f2bf(float f) {
    unsigned short s = f2bfu(f);
    return __builtin_bit_cast(__bf16, s);
}
__device__ __forceinline__ float b2f(__bf16 h) {
    unsigned short s = __builtin_bit_cast(unsigned short, h);
    unsigned u = ((unsigned)s) << 16;
    return __builtin_bit_cast(float, u);
}

// Async copy: 16 bytes global -> LDS, tracked by ASYNCcnt (per-lane addresses,
// so it also implements the partition gather). CDNA5-specific data path.
__device__ __forceinline__ void async_b128(unsigned lds_off, const void* gaddr) {
    asm volatile("global_load_async_to_lds_b128 %0, %1, off"
                 :: "v"(lds_off), "v"(gaddr) : "memory");
}
__device__ __forceinline__ unsigned lds_off(const void* p) {
    return (unsigned)(uintptr_t)p;   // flat LDS aperture: addr[31:0] = LDS offset
}
__device__ __forceinline__ void wait_async_0() {
    asm volatile("s_wait_asynccnt 0x0" ::: "memory");
}
__device__ __forceinline__ void wait_async_4() {
    asm volatile("s_wait_asynccnt 0x4" ::: "memory");
}

// A fragment, 16x32 (MxK), bf16: lane 0-15 -> M, lanes 16-31 -> same M, upper K half.
__device__ __forceinline__ v16bf ldA(const __bf16* S, int ld, int m0, int k0, int lane) {
    int m  = m0 + (lane & 15);
    int kh = (lane & 16) ? 8 : 0;
    v16bf a;
#pragma unroll
    for (int h = 0; h < 16; ++h) {
        int kk = ((h < 8) ? h : h + 8) + kh;
        a[h] = S[m * ld + k0 + kk];
    }
    return a;
}
// A fragment with transpose: logical A[m][k] = S[k][m]
__device__ __forceinline__ v16bf ldAT(const __bf16* S, int ld, int m0, int k0, int lane) {
    int m  = m0 + (lane & 15);
    int kh = (lane & 16) ? 8 : 0;
    v16bf a;
#pragma unroll
    for (int h = 0; h < 16; ++h) {
        int kk = ((h < 8) ? h : h + 8) + kh;
        a[h] = S[(k0 + kk) * ld + m];
    }
    return a;
}
// B fragment, 32x16 (KxN): lane 0-15 -> N, lanes 16-31 -> same N, upper K half (same K order as A).
__device__ __forceinline__ v16bf ldB(const __bf16* S, int ld, int k0, int n0, int lane) {
    int n  = n0 + (lane & 15);
    int kh = (lane & 16) ? 8 : 0;
    v16bf b;
#pragma unroll
    for (int h = 0; h < 16; ++h) b[h] = S[(k0 + ((h < 8) ? h : h + 8) + kh) * ld + n];
    return b;
}
// B fragment with transpose: logical B[k][n] = S[n][k]
__device__ __forceinline__ v16bf ldBT(const __bf16* S, int ld, int k0, int n0, int lane) {
    int n  = n0 + (lane & 15);
    int kh = (lane & 16) ? 8 : 0;
    v16bf b;
#pragma unroll
    for (int h = 0; h < 16; ++h) b[h] = S[n * ld + k0 + ((h < 8) ? h : h + 8) + kh];
    return b;
}
__device__ __forceinline__ v8f wmma_bf16(v16bf a, v16bf b, v8f c) {
    return __builtin_amdgcn_wmma_f32_16x16x32_bf16(false, a, false, b, (short)0, c, false, false);
}

// ---------------------------------------------------------------------------
// Kernel 0: zero small reduction area (2 + 256 + 16384 floats)
// ---------------------------------------------------------------------------
__global__ __launch_bounds__(256) void k_zero(float* red) {
    int i = blockIdx.x * 256 + threadIdx.x;
    if (i < 2 + 256 + 4096 * NHEAD) red[i] = 0.0f;
}

// ---------------------------------------------------------------------------
// Kernel 1: one-time casts: x -> bf16, W -> bf16 pre-transposed [mat][k][oc]
// ---------------------------------------------------------------------------
__global__ __launch_bounds__(256) void k_cast(const float* __restrict__ x,
                                              const float* __restrict__ Wq,
                                              const float* __restrict__ Wk,
                                              const float* __restrict__ Wv,
                                              __bf16* __restrict__ xh,
                                              __bf16* __restrict__ WhT) {
    size_t gid = (size_t)blockIdx.x * 256 + threadIdx.x;
    size_t stride = (size_t)gridDim.x * 256;
    for (size_t i = gid; i < (size_t)NROWS * INC; i += stride) xh[i] = f2bf(x[i]);
    for (size_t i = gid; i < 3u * 65536u; i += stride) {
        int mat = (int)(i >> 16);
        int r = (int)((i >> 8) & 255);     // k index
        int c = (int)(i & 255);            // output-channel index
        const float* W = (mat == 0) ? Wq : (mat == 1) ? Wk : Wv;
        WhT[i] = f2bf(W[c * 256 + r]);     // WhT[mat][k][oc] = W[oc][k]
    }
}

// ---------------------------------------------------------------------------
// Kernel 2: QKV projection  Y = x @ W^T + b   (bf16 WMMA, f32 accumulate)
// Double-buffered async-to-LDS pipeline; bf16 outputs Qh/Kh/Vh.
// grid: (N/128, 6); blockIdx.y selects matrix (Q/K/V) and 128-col half
// ---------------------------------------------------------------------------
__global__ __launch_bounds__(256) void k_qkv(const __bf16* __restrict__ xh,
                                             const __bf16* __restrict__ WhT,
                                             const float* __restrict__ bq,
                                             const float* __restrict__ bk,
                                             const float* __restrict__ bv,
                                             __bf16* __restrict__ Qh,
                                             __bf16* __restrict__ Kh,
                                             __bf16* __restrict__ Vh) {
    __shared__ alignas(16) __bf16 As[2][128 * 32];   // [row][k]
    __shared__ alignas(16) __bf16 Bs[2][32 * 128];   // [k][out-col]
    const int tid = threadIdx.x, lane = tid & 31, wid = tid >> 5;
    const int row0 = blockIdx.x * 128;
    const int gy = blockIdx.y, mat = gy >> 1, col0 = (gy & 1) * 128;
    const __bf16* Wm   = WhT + (size_t)mat * 65536;
    const float*  bias = (mat == 0) ? bq : (mat == 1) ? bk : bv;
    __bf16*       Outh = (mat == 0) ? Qh : (mat == 1) ? Kh : Vh;
    const int wm = wid & 3, wn = wid >> 2;   // 4x2 wave grid: 32 rows x 64 cols per wave

    auto issue = [&](int k0, int buf) {
        // A tile: 128 rows x 32 bf16 = 512 x 16B segments (2 per thread)
#pragma unroll
        for (int e = tid; e < 512; e += 256) {
            int r = e >> 2, s4 = e & 3;
            async_b128(lds_off(&As[buf][r * 32 + s4 * 8]),
                       xh + (size_t)(row0 + r) * INC + k0 + s4 * 8);
        }
        // B tile: 32 k x 128 cols = 512 x 16B segments (2 per thread)
#pragma unroll
        for (int e = tid; e < 512; e += 256) {
            int kk = e >> 4, s16 = e & 15;
            async_b128(lds_off(&Bs[buf][kk * 128 + s16 * 8]),
                       Wm + (size_t)(k0 + kk) * INC + col0 + s16 * 8);
        }
    };

    v8f zero = {};
    v8f acc[2][4];
#pragma unroll
    for (int a = 0; a < 2; ++a)
#pragma unroll
        for (int b = 0; b < 4; ++b) acc[a][b] = zero;

    issue(0, 0);
    for (int ks = 0; ks < 8; ++ks) {
        int buf = ks & 1;
        if (ks < 7) { issue((ks + 1) * 32, buf ^ 1); wait_async_4(); }
        else        { wait_async_0(); }
        __syncthreads();                           // stage ks resident for all waves
        const __bf16* Asb = As[buf];
        const __bf16* Bsb = Bs[buf];
        v16bf a0 = ldA(Asb, 32, wm * 32,      0, lane);
        v16bf a1 = ldA(Asb, 32, wm * 32 + 16, 0, lane);
#pragma unroll
        for (int nt = 0; nt < 4; ++nt) {
            v16bf b = ldB(Bsb, 128, 0, wn * 64 + nt * 16, lane);
            acc[0][nt] = wmma_bf16(a0, b, acc[0][nt]);
            acc[1][nt] = wmma_bf16(a1, b, acc[1][nt]);
        }
        __syncthreads();                           // done reading before buffer reuse
    }
    // epilogue: add bias, write bf16
    const int mrowofs = (lane & 16) ? 8 : 0;
#pragma unroll
    for (int mt = 0; mt < 2; ++mt)
#pragma unroll
        for (int nt = 0; nt < 4; ++nt)
#pragma unroll
            for (int i = 0; i < 8; ++i) {
                int r = row0 + wm * 32 + mt * 16 + i + mrowofs;
                int c = col0 + wn * 64 + nt * 16 + (lane & 15);
                Outh[(size_t)r * INC + c] = f2bf(acc[mt][nt][i] + bias[c]);
            }
}

// ---------------------------------------------------------------------------
// Kernel 3: reductions: sum(Q^2), sum(K^2), column-sums of K, K^T V per head
// grid: (N/128, NHEAD)
// ---------------------------------------------------------------------------
__global__ __launch_bounds__(256) void k_reduce(const __bf16* __restrict__ Qh,
                                                const __bf16* __restrict__ Kh,
                                                const __bf16* __restrict__ Vh,
                                                float* __restrict__ red,
                                                float* __restrict__ ksum,
                                                float* __restrict__ kvraw) {
    __shared__ alignas(16) __bf16 Kb[128 * 64];
    __shared__ alignas(16) __bf16 Vb[128 * 64];
    __shared__ float sred[256];
    const int tid = threadIdx.x, lane = tid & 31, wid = tid >> 5;
    const int r0 = blockIdx.x * 128, h = blockIdx.y;

    // stage K/V chunk via async copies (128 rows x 64 bf16 = 1024 segs each)
#pragma unroll
    for (int e = tid; e < 1024; e += 256) {
        int s = e >> 3, seg = e & 7;
        size_t gb = (size_t)(r0 + s) * INC + h * DHEAD + seg * 8;
        async_b128(lds_off(&Kb[s * 64 + seg * 8]), Kh + gb);
        async_b128(lds_off(&Vb[s * 64 + seg * 8]), Vh + gb);
    }
    // overlap: Q sum-of-squares from global while copies fly
    float sqq = 0.f;
    for (int e = tid; e < 128 * 64; e += 256) {
        int r = e >> 6, c = e & 63;
        float qv = b2f(Qh[(size_t)(r0 + r) * INC + h * DHEAD + c]);
        sqq += qv * qv;
    }
    wait_async_0();
    __syncthreads();

    float sqk = 0.f;
    for (int e = tid; e < 128 * 64; e += 256) { float kv = b2f(Kb[e]); sqk += kv * kv; }

    // block reductions
    sred[tid] = sqq; __syncthreads();
    for (int s = 128; s > 0; s >>= 1) { if (tid < s) sred[tid] += sred[tid + s]; __syncthreads(); }
    if (tid == 0) atomicAdd(&red[0], sred[0]);
    __syncthreads();
    sred[tid] = sqk; __syncthreads();
    for (int s = 128; s > 0; s >>= 1) { if (tid < s) sred[tid] += sred[tid + s]; __syncthreads(); }
    if (tid == 0) atomicAdd(&red[1], sred[0]);
    __syncthreads();

    // column sums of K chunk
    if (tid < 64) {
        float s = 0.f;
        for (int r = 0; r < 128; ++r) s += b2f(Kb[r * 64 + tid]);
        atomicAdd(&ksum[h * DHEAD + tid], s);
    }
    // K^T V (64x64, reduce over 128 rows): waves 4x2, 2 C-tiles each
    const int wm = wid >> 1, wn2 = (wid & 1) * 2;
    v8f z = {}; v8f c0 = z, c1 = z;
#pragma unroll
    for (int ks = 0; ks < 4; ++ks) {
        int k0 = ks * 32;
        v16bf a  = ldAT(Kb, 64, wm * 16, k0, lane);     // A[dk][row] = K[row][dk]
        v16bf b0 = ldB (Vb, 64, k0, wn2 * 16,       lane);
        v16bf b1 = ldB (Vb, 64, k0, (wn2 + 1) * 16, lane);
        c0 = wmma_bf16(a, b0, c0);
        c1 = wmma_bf16(a, b1, c1);
    }
    const int mrowofs = (lane & 16) ? 8 : 0;
#pragma unroll
    for (int i = 0; i < 8; ++i) {
        int m = wm * 16 + i + mrowofs;
        int n = wn2 * 16 + (lane & 15);
        atomicAdd(&kvraw[h * 4096 + m * DHEAD + n],      c0[i]);
        atomicAdd(&kvraw[h * 4096 + m * DHEAD + n + 16], c1[i]);
    }
}

// ---------------------------------------------------------------------------
// Kernel 4: fold norms into kvs2 (bf16) and ksnn
// ---------------------------------------------------------------------------
__global__ __launch_bounds__(256) void k_prep(const float* __restrict__ red,
                                              float* __restrict__ ksnn,
                                              __bf16* __restrict__ kvs2) {
    const float* ksum  = red + 2;
    const float* kvraw = red + 2 + 256;
    float inv = rsqrtf(red[0]) * rsqrtf(red[1]);      // 1 / (||Q|| * ||K||)
    int gid = blockIdx.x * 256 + threadIdx.x;
    for (int e = gid; e < 4096 * NHEAD; e += gridDim.x * 256)
        kvs2[e] = f2bf(kvraw[e] * inv);
    if (gid < NHEAD * DHEAD) ksnn[gid] = ksum[gid] * inv;
}

// ---------------------------------------------------------------------------
// Kernel 5: per-partition softmax attention + global linear-attention combine
// grid: (256). One workgroup per partition, ~233 KB dynamic LDS.
// ---------------------------------------------------------------------------
__global__ __launch_bounds__(256) void k_attn(const __bf16* __restrict__ Qh,
                                              const __bf16* __restrict__ Kh,
                                              const __bf16* __restrict__ Vh,
                                              const int* __restrict__ idx,
                                              const float* __restrict__ ksnn,
                                              const __bf16* __restrict__ kvs2,
                                              const float* __restrict__ gamma_logit,
                                              float* __restrict__ out) {
    extern __shared__ char smem[];
    __bf16* Qb  = (__bf16*)smem;        // 256x64
    __bf16* Kb  = Qb + 256 * 64;        // 256x64
    __bf16* Vb  = Kb + 256 * 64;        // 256x64
    __bf16* Sb  = Vb + 256 * 64;        // 256x256 scores/probs
    __bf16* kb2 = Sb + 256 * 256;       // 64x64 kvs2[h]
    float*  denomv = (float*)(kb2 + 64 * 64);   // 256

    const int tid = threadIdx.x, lane = tid & 31, wid = tid >> 5;
    const int p = blockIdx.x;
    const int* pidx = idx + p * 256;
    const float g  = 1.0f / (1.0f + __expf(-gamma_logit[0]));
    const float og = 1.0f - g;
    const float nN = (float)NROWS;
    const int m0 = wid * 32;                   // 32 output rows per wave
    const int mrowofs = (lane & 16) ? 8 : 0;

    v8f zero = {};
    v8f accum[2][4];
#pragma unroll
    for (int a = 0; a < 2; ++a)
#pragma unroll
        for (int b = 0; b < 4; ++b) accum[a][b] = zero;

    for (int h = 0; h < NHEAD; ++h) {
        __syncthreads();                        // protect LDS reuse across heads
        // gather Q/K/V rows via per-lane async b128 copies (256 rows x 8 segs)
        for (int e = tid; e < 2048; e += 256) {
            int s = e >> 3, seg = e & 7;
            size_t gb = (size_t)pidx[s] * INC + h * DHEAD + seg * 8;
            unsigned lofs = (unsigned)((s * 64 + seg * 8) * 2);
            async_b128(lds_off(Qb) + lofs, Qh + gb);
            async_b128(lds_off(Kb) + lofs, Kh + gb);
            async_b128(lds_off(Vb) + lofs, Vh + gb);
        }
        // kvs2 head tile: 4096 bf16 = 512 segs
#pragma unroll
        for (int e = tid; e < 512; e += 256)
            async_b128(lds_off(&kb2[e * 8]), kvs2 + (size_t)h * 4096 + e * 8);
        wait_async_0();
        __syncthreads();

        // denom[s] = q_s . ksnn + n   (one row per thread)
        {
            float d = 0.f;
#pragma unroll
            for (int c = 0; c < DHEAD; ++c) d += b2f(Qb[tid * 64 + c]) * ksnn[h * DHEAD + c];
            denomv[tid] = d + nN;
        }

        // scores S = (Q K^T) / sqrt(D): wave handles rows [m0, m0+32) x all 256 cols
        for (int nt = 0; nt < 16; ++nt) {
            v8f s0 = zero, s1 = zero;
#pragma unroll
            for (int ks = 0; ks < 2; ++ks) {
                int k0 = ks * 32;
                v16bf a0 = ldA (Qb, 64, m0,      k0, lane);
                v16bf a1 = ldA (Qb, 64, m0 + 16, k0, lane);
                v16bf b  = ldBT(Kb, 64, k0, nt * 16, lane);   // B[d][t] = K[t][d]
                s0 = wmma_bf16(a0, b, s0);
                s1 = wmma_bf16(a1, b, s1);
            }
            int n = nt * 16 + (lane & 15);
#pragma unroll
            for (int i = 0; i < 8; ++i) {
                Sb[(m0 + i + mrowofs)      * 256 + n] = f2bf(s0[i] * 0.125f);
                Sb[(m0 + 16 + i + mrowofs) * 256 + n] = f2bf(s1[i] * 0.125f);
            }
        }
        __syncthreads();

        // softmax over each row (one row per thread)
        {
            float mx = -1e30f;
            for (int t = 0; t < 256; ++t) mx = fmaxf(mx, b2f(Sb[tid * 256 + t]));
            float sum = 0.f;
            for (int t = 0; t < 256; ++t) sum += __expf(b2f(Sb[tid * 256 + t]) - mx);
            float inv = 1.0f / sum;
            for (int t = 0; t < 256; ++t)
                Sb[tid * 256 + t] = f2bf(__expf(b2f(Sb[tid * 256 + t]) - mx) * inv);
        }
        __syncthreads();

        // O = P @ V   (K-dim 256)
        v8f o[2][4];
#pragma unroll
        for (int a = 0; a < 2; ++a)
#pragma unroll
            for (int b = 0; b < 4; ++b) o[a][b] = zero;
        for (int ks = 0; ks < 8; ++ks) {
            int k0 = ks * 32;
            v16bf a0 = ldA(Sb, 256, m0,      k0, lane);
            v16bf a1 = ldA(Sb, 256, m0 + 16, k0, lane);
#pragma unroll
            for (int nt = 0; nt < 4; ++nt) {
                v16bf b = ldB(Vb, 64, k0, nt * 16, lane);
                o[0][nt] = wmma_bf16(a0, b, o[0][nt]);
                o[1][nt] = wmma_bf16(a1, b, o[1][nt]);
            }
        }
        // G = Q @ kvs2[h]   (K-dim 64)
        v8f gq[2][4];
#pragma unroll
        for (int a = 0; a < 2; ++a)
#pragma unroll
            for (int b = 0; b < 4; ++b) gq[a][b] = zero;
#pragma unroll
        for (int ks = 0; ks < 2; ++ks) {
            int k0 = ks * 32;
            v16bf a0 = ldA(Qb, 64, m0,      k0, lane);
            v16bf a1 = ldA(Qb, 64, m0 + 16, k0, lane);
#pragma unroll
            for (int nt = 0; nt < 4; ++nt) {
                v16bf b = ldB(kb2, 64, k0, nt * 16, lane);
                gq[0][nt] = wmma_bf16(a0, b, gq[0][nt]);
                gq[1][nt] = wmma_bf16(a1, b, gq[1][nt]);
            }
        }
        // fused epilogue: mean over heads of g*attn + (1-g)*(num + n*V)/denom
#pragma unroll
        for (int mt = 0; mt < 2; ++mt)
#pragma unroll
            for (int nt = 0; nt < 4; ++nt)
#pragma unroll
                for (int i = 0; i < 8; ++i) {
                    int s = m0 + mt * 16 + i + mrowofs;
                    int c = nt * 16 + (lane & 15);
                    float vf  = b2f(Vb[s * 64 + c]);      // V from resident LDS tile
                    float num = gq[mt][nt][i] + nN * vf;
                    float val = g * o[mt][nt][i] + og * num / denomv[s];
                    accum[mt][nt][i] += 0.25f * val;
                }
    }
    // write output [N, 64]
#pragma unroll
    for (int mt = 0; mt < 2; ++mt)
#pragma unroll
        for (int nt = 0; nt < 4; ++nt)
#pragma unroll
            for (int i = 0; i < 8; ++i) {
                int s = m0 + mt * 16 + i + mrowofs;
                int c = nt * 16 + (lane & 15);
                out[(size_t)pidx[s] * DHEAD + c] = accum[mt][nt][i];
            }
}

// ---------------------------------------------------------------------------
// Host entry
// ---------------------------------------------------------------------------
extern "C" void kernel_launch(void* const* d_in, const int* in_sizes, int n_in,
                              void* d_out, int out_size, void* d_ws, size_t ws_size,
                              hipStream_t stream) {
    const float* x     = (const float*)d_in[0];
    const int*   idx   = (const int*)  d_in[1];
    const float* Wq    = (const float*)d_in[2];
    const float* bq    = (const float*)d_in[3];
    const float* Wk    = (const float*)d_in[4];
    const float* bk    = (const float*)d_in[5];
    const float* Wv    = (const float*)d_in[6];
    const float* bv    = (const float*)d_in[7];
    const float* gamma = (const float*)d_in[8];
    float* out = (float*)d_out;

    // workspace layout (all bf16 tensors 16B-aligned for async b128 copies)
    __bf16* xh  = (__bf16*)d_ws;                       // N*256
    __bf16* Qh  = xh + (size_t)NROWS * INC;            // N*256
    __bf16* Kh  = Qh + (size_t)NROWS * INC;
    __bf16* Vh  = Kh + (size_t)NROWS * INC;
    __bf16* WhT = Vh + (size_t)NROWS * INC;            // 3*256*256
    float*  red = (float*)(WhT + 3 * 65536);           // [0]=sumsqQ,[1]=sumsqK
    float*  ksum  = red + 2;                           // 256
    float*  kvraw = red + 2 + 256;                     // 4*64*64
    float*  ksnn  = red + 2 + 256 + 4096 * NHEAD;      // 256
    __bf16* kvs2  = (__bf16*)(red + 16904);            // 4*64*64 bf16 (16B aligned)

    k_zero  <<<66, 256, 0, stream>>>(red);
    k_cast  <<<1024, 256, 0, stream>>>(x, Wq, Wk, Wv, xh, WhT);
    k_qkv   <<<dim3(NROWS / 128, 6), 256, 0, stream>>>(xh, WhT, bq, bk, bv, Qh, Kh, Vh);
    k_reduce<<<dim3(NROWS / 128, NHEAD), 256, 0, stream>>>(Qh, Kh, Vh, red, ksum, kvraw);
    k_prep  <<<16, 256, 0, stream>>>(red, ksnn, kvs2);

    // dynamic LDS: 3*(256*64) + 256*256 + 64*64 bf16 elems + 256 floats = 238592 B
    size_t lds = (size_t)(3 * 256 * 64 + 256 * 256 + 64 * 64) * 2 + 256 * 4;
    k_attn  <<<256, 256, lds, stream>>>(Qh, Kh, Vh, idx, ksnn, kvs2, gamma, out);
}
